// S4__52604759441727
// MI455X (gfx1250) — matrix operand order repaired and checked by reading, not compiled, not run on hardware
//
#include <hip/hip_runtime.h>

// ---- problem dims (fixed by the reference) ----
#define TT   354      // timesteps
#define HH   128      // model dim
#define NSM  32       // SSM modes
#define NB   16       // graphs
#define NN   64       // nodes
#define BNN  (NB*NN)  // 1024 independent scalar series
#define TPK  384      // T padded to a multiple of 32 (K buffers, zero-filled tail)
#define LT_TILES 23   // ceil(354/16) l-tiles of 16 rows
#define SRLEN 432     // reversed-series copy length (covers k in [0,423] incl. shifts)

typedef __attribute__((ext_vector_type(16))) __bf16 v16bf;
typedef __attribute__((ext_vector_type(8)))  float  v8f;
typedef __attribute__((ext_vector_type(4)))  unsigned int u32x4;
typedef __attribute__((ext_vector_type(8)))  int i32x8;
typedef __attribute__((ext_vector_type(4)))  int i32x4;

union FragAB { v16bf v; unsigned short u[16]; uint4 q[2]; };
union FragC  { v8f   v; float f[8]; };

__device__ __forceinline__ unsigned short f2bf(float x) {       // f32 -> bf16 RNE
  unsigned int u = __builtin_bit_cast(unsigned int, x);
  u = (u + 0x7FFFu + ((u >> 16) & 1u)) >> 16;
  return (unsigned short)u;
}
__device__ __forceinline__ float gelu_tanh(float x) {           // jax.nn.gelu (approximate)
  return 0.5f * x * (1.0f + tanhf(0.7978845608028654f * (x + 0.044715f * x * x * x)));
}
__device__ __forceinline__ float sigmoidf(float x) { return 1.0f / (1.0f + __expf(-x)); }

// ============================================================================
// Kernel 1: build S4D conv kernel K (bf16, K-major [h][TPK] for WMMA-B loads)
// and the analytic prefix sum Ksum[l][h] = 2*Re( sum_n Cd * (1-w^{l+1})/(1-w) )
// (needed for the b_in term after factoring u = W_in*s + b_in out of the conv).
// ============================================================================
__global__ __launch_bounds__(256) void s4_prep_kernel(
    const float* __restrict__ log_dt, const float* __restrict__ A_re,
    const float* __restrict__ A_im,  const float* __restrict__ C_re,
    const float* __restrict__ C_im,
    unsigned short* __restrict__ KtT,   // [HH][TPK] bf16
    float* __restrict__ Ksum)           // [TPK][HH] f32
{
  int idx = blockIdx.x * 256 + threadIdx.x;
  if (idx >= TPK * HH) return;
  int h = idx % HH, t = idx / HH;
  if (t >= TT) { KtT[h * TPK + t] = 0; Ksum[(size_t)t * HH + h] = 0.f; return; }

  float dt = __expf(log_dt[h]);
  float kacc = 0.f, sacc = 0.f;
  for (int n = 0; n < NSM; ++n) {
    float ar = A_re[h * NSM + n], ai = A_im[h * NSM + n];
    float cr = C_re[h * NSM + n], ci = C_im[h * NSM + n];
    float dar = ar * dt, dai = ai * dt;
    // w = exp(dt*A)
    float er = __expf(dar), sw, cw; __sincosf(dai, &sw, &cw);
    float wr = er * cw, wi = er * sw;
    // Cd = C*(w-1)/A
    float t1r = cr * (wr - 1.f) - ci * wi;
    float t1i = cr * wi + ci * (wr - 1.f);
    float den = ar * ar + ai * ai;
    float cdr = (t1r * ar + t1i * ai) / den;
    float cdi = (t1i * ar - t1r * ai) / den;
    // w^t
    float et = __expf(dar * (float)t), st, ct; __sincosf(dai * (float)t, &st, &ct);
    float wtr = et * ct, wti = et * st;
    kacc += 2.f * (cdr * wtr - cdi * wti);
    // prefix sum: (1 - w^{t+1}) / (1 - w)
    float w1r = wtr * wr - wti * wi, w1i = wtr * wi + wti * wr;
    float nr = 1.f - w1r, ni = -w1i;
    float dr = 1.f - wr,  di = -wi;
    float dd = dr * dr + di * di;
    float srr = (nr * dr + ni * di) / dd;
    float sri = (ni * dr - nr * di) / dd;
    sacc += 2.f * (cdr * srr - cdi * sri);
  }
  KtT[h * TPK + t] = f2bf(kacc);
  Ksum[(size_t)t * HH + h] = sacc;
}

// ============================================================================
// Kernel 2 (fused): one workgroup (8 wave32s) per scalar series bn.
//  stage  : per-node strided series gathered into LDS by the Tensor Data Mover
//           (2D D#: tile 1 x 354, stride 64 elems), s_wait_tensorcnt + barrier
//  phase 1: cs[l,h] = causal Toeplitz(s) x K via WMMA; Toeplitz A-fragments are
//           branchless aligned ds_load_b128 from a shift-replicated reversed
//           series (srev8[s][k] = s[383-(k+s)], zero-padded outside [0,T)).
//  phase 2: y = W_in*cs + b_in*Ksum + D*u; GELU; y-tile (16x128 bf16) -> LDS
//  phase 3: g = y @ W_out^T (2H cols) via WMMA, W_out frags persistent in VGPRs
//  phase 4: GLU + masked time-sum accumulation -> node_sum[bn][h]
// ============================================================================
__global__ __launch_bounds__(256) void s4_main_kernel(
    const float* __restrict__ x,
    const float* __restrict__ W_in,  const float* __restrict__ b_in,
    const float* __restrict__ Dskip,
    const float* __restrict__ W_out, const float* __restrict__ b_out,
    const unsigned short* __restrict__ KtT,
    const float* __restrict__ Ksum,
    float* __restrict__ node_sum)
{
  __shared__ float          s_f32[TPK];       // series, f32 (skip path)
  __shared__ unsigned short s_bf[TPK];        // series, bf16 (zero-padded tail)
  __shared__ unsigned short srev8[8 * SRLEN]; // 8 shifted copies of reversed series
  __shared__ unsigned short ytile[16 * HH];   // one 16 x 128 y-tile, bf16
  __shared__ float          red[256];

  const int bn = blockIdx.x;
  const int b = bn / NN, nnode = bn % NN;
  const int tid = threadIdx.x;
  const int w    = tid >> 5;     // wave 0..7 (wave32)
  const int lane = tid & 31;
  const int hk   = lane >> 4;    // lane half (ISA 16-bit operand layout)
  const int ln   = lane & 15;

  // ---- stage the per-node scalar series (x is (B,T,N): stride-64 column) ----
#if __has_builtin(__builtin_amdgcn_tensor_load_to_lds) && __has_builtin(__builtin_amdgcn_s_wait_tensorcnt)
  if (tid < TPK - TT) s_f32[TT + tid] = 0.f;    // zero tail (threads 0..29, wave 0)
  if (w == 0) {                                  // TDM ignores EXEC: issue from one wave
    unsigned long long ga =
        (unsigned long long)(uintptr_t)(x + (size_t)b * TT * NN + nnode);
    unsigned int lds_addr = (unsigned int)(uintptr_t)&s_f32[0];
    // D# group 0: count=1, lds_addr[63:32], global_addr[120:64], type=2
    u32x4 g0 = { 1u, lds_addr,
                 (unsigned int)(ga & 0xFFFFFFFFu),
                 (unsigned int)(((ga >> 32) & 0x01FFFFFFu) | (2u << 30)) };
    // D# group 1: data_size=4B (code 2), tensor_dim0=NN, tensor_dim1=TT,
    //             tile_dim0=1, tile_dim1=TT, tensor_dim0_stride=NN
    i32x8 g1 = { (int)0x20000,        // [17:16] data_size=2
                 (int)(NN << 16),     // tensor_dim0 low16 -> bits[63:48]
                 (int)(TT << 16),     // tensor_dim0 hi16=0, tensor_dim1 low16 -> bits[95:80]
                 (int)(1  << 16),     // tensor_dim1 hi16=0, tile_dim0=1 -> bits[127:112]
                 (int)TT,             // tile_dim1 -> bits[143:128]
                 (int)NN,             // tensor_dim0_stride low32 -> bits[191:160]
                 0, 0 };
    i32x4 gz4 = { 0, 0, 0, 0 };             // 2D tensor: groups 2/3 unused
    i32x8 gz8 = { 0, 0, 0, 0, 0, 0, 0, 0 }; // trailing group (6-arg toolchain form)
    __builtin_amdgcn_tensor_load_to_lds(g0, g1, gz4, gz4, gz8, 0);
    __builtin_amdgcn_s_wait_tensorcnt(0);
  }
  __syncthreads();
  for (int t = tid; t < TPK; t += 256) s_bf[t] = f2bf(s_f32[t]);
#else
  for (int t = tid; t < TPK; t += 256) {
    float v = (t < TT) ? x[(size_t)b * TT * NN + (size_t)t * NN + nnode] : 0.f;
    s_f32[t] = v;
    s_bf[t]  = f2bf(v);
  }
#endif
  __syncthreads();

  // build shift-replicated reversed series: srev8[s*SRLEN + k] = sval(383-(k+s))
  for (int k = tid; k < 8 * SRLEN; k += 256) {
    int row = k / SRLEN;
    int kk  = k - row * SRLEN;
    int t   = 383 - (kk + row);               // t <= 383 always
    srev8[k] = (t >= 0) ? s_bf[t] : (unsigned short)0;
  }

  // per-lane channel constants: conv h-column and GLU output column are both 16*w+ln
  const int hC = 16 * w + ln;
  const float win_h = W_in[hC];
  const float bin_h = b_in[hC];
  const float dsk_h = Dskip[hC];
  const float bo_v  = b_out[hC];        // value-half bias
  const float bo_g  = b_out[hC + HH];   // gate-half bias

  // persistent W_out B-fragments (K = model dim h, N = output col g).
  // B layout: lane -> N = ln, element e -> K = e + 16*hk (contiguous).
  FragAB wof[8];
  for (int half = 0; half < 2; ++half) {
    int g = (half ? (16 * w + HH) : (16 * w)) + ln;
    for (int kc = 0; kc < 4; ++kc) {
      FragAB f;
#pragma unroll
      for (int e = 0; e < 16; ++e)
        f.u[e] = f2bf(W_out[(size_t)g * HH + kc * 32 + e + 16 * hk]);
      wof[half * 4 + kc] = f;
    }
  }

  // per-lane reversed-series addressing:
  //  base(lt,jc) = 383 - l0 - ln + j0 + 16*hk ; shift sh = base & 7 = (7-ln)&7
  //  aligned element index ab = base - sh  -> 16B-aligned ds_load_b128
  const int sh = (7 - ln) & 7;
  const unsigned short* srow = srev8 + sh * SRLEN;   // srow[k] = srev[k+sh]
  const int cbase = 383 - ln + 16 * hk - sh;         // ab = cbase - l0 + j0

  __syncthreads();

  float zsum = 0.f;  // sum over time of z for column hC

  for (int lt = 0; lt < LT_TILES; ++lt) {
    const int l0 = lt * 16;

    // ---- phase 1: conv tile, rows l0..l0+15, cols 16w..16w+15 ----
    FragC acc;
#pragma unroll
    for (int r = 0; r < 8; ++r) acc.f[r] = 0.f;
    const int nchunks = (l0 + 15) / 32 + 1;          // causal: j <= l
    const int ab0 = cbase - l0;
    const unsigned short* kcol = KtT + (size_t)hC * TPK + 16 * hk;
    for (int jc = 0; jc < nchunks; ++jc) {
      const int j0 = jc * 32;
      // A (Toeplitz): two aligned b128 loads, branchless
      FragAB fa;
      fa.q[0] = *(const uint4*)(srow + (ab0 + j0));
      fa.q[1] = *(const uint4*)(srow + (ab0 + j0 + 8));
      // B (K matrix): 16 contiguous bf16, 32B aligned; WGP-scope prefetch of next chunk
      const v16bf fb = *(const v16bf*)(kcol + j0);
      __builtin_prefetch(kcol + j0 + 32, 0, 3);
      acc.v = __builtin_amdgcn_wmma_f32_16x16x32_bf16(
          false, fa.v, false, fb, (short)0, acc.v, false, false);
    }

    // ---- phase 2: pointwise + GELU -> y-tile in LDS ----
#pragma unroll
    for (int r = 0; r < 8; ++r) {
      int m = r + 8 * hk;            // C layout: M = r + 8*(lane>=16), N = ln
      int l = l0 + m;
      float u_l = win_h * s_f32[l] + bin_h;
      float yv  = win_h * acc.f[r] + bin_h * Ksum[(size_t)l * HH + hC] + dsk_h * u_l;
      ytile[m * HH + hC] = f2bf(gelu_tanh(yv));
    }
    __syncthreads();

    // ---- phase 3: g = y @ W_out^T for this wave's (value, gate) column tiles ----
    FragC g0f, g1f;
#pragma unroll
    for (int r = 0; r < 8; ++r) { g0f.f[r] = 0.f; g1f.f[r] = 0.f; }
#pragma unroll
    for (int kc = 0; kc < 4; ++kc) {
      FragAB fa;   // A rows = ln; two aligned b128 LDS loads per K-chunk
      fa.q[0] = *(const uint4*)(ytile + ln * HH + kc * 32 + 16 * hk);
      fa.q[1] = *(const uint4*)(ytile + ln * HH + kc * 32 + 8 + 16 * hk);
      g0f.v = __builtin_amdgcn_wmma_f32_16x16x32_bf16(
          false, fa.v, false, wof[kc].v,     (short)0, g0f.v, false, false);
      g1f.v = __builtin_amdgcn_wmma_f32_16x16x32_bf16(
          false, fa.v, false, wof[4 + kc].v, (short)0, g1f.v, false, false);
    }

    // ---- phase 4: GLU + masked time accumulation (branchless select) ----
#pragma unroll
    for (int r = 0; r < 8; ++r) {
      int l = l0 + r + 8 * hk;
      float msk = (l < TT) ? 1.f : 0.f;
      float gv = g0f.f[r] + bo_v;
      float gg = g1f.f[r] + bo_g;
      zsum += msk * gv * sigmoidf(gg);
    }
    __syncthreads();   // ytile rewritten next iteration
  }

  // combine lane halves (L and L+16 hold the same output column hC)
  red[tid] = zsum;
  __syncthreads();
  if (hk == 0)
    node_sum[(size_t)bn * HH + hC] = red[tid] + red[tid + 16];
}

// ============================================================================
// Kernel 3: node mean + classifier. out[b] = mean_{n,l}(z) . W_cls + b_cls
// ============================================================================
__global__ __launch_bounds__(128) void s4_final_kernel(
    const float* __restrict__ node_sum, const float* __restrict__ W_cls,
    const float* __restrict__ b_cls, float* __restrict__ out)
{
  __shared__ float red[128];
  int b = blockIdx.x, h = threadIdx.x;
  float acc = 0.f;
  for (int n = 0; n < NN; ++n)
    acc += node_sum[(size_t)(b * NN + n) * HH + h];
  acc *= W_cls[h] * (1.0f / ((float)NN * (float)TT));
  red[h] = acc;
  __syncthreads();
  for (int s = 64; s > 0; s >>= 1) {
    if (h < s) red[h] += red[h + s];
    __syncthreads();
  }
  if (h == 0) out[b] = red[0] + b_cls[0];
}

// ============================================================================
extern "C" void kernel_launch(void* const* d_in, const int* in_sizes, int n_in,
                              void* d_out, int out_size, void* d_ws, size_t ws_size,
                              hipStream_t stream) {
  const float* x      = (const float*)d_in[0];
  const float* W_in   = (const float*)d_in[1];
  const float* b_in   = (const float*)d_in[2];
  const float* log_dt = (const float*)d_in[3];
  const float* A_re   = (const float*)d_in[4];
  const float* A_im   = (const float*)d_in[5];
  const float* C_re   = (const float*)d_in[6];
  const float* C_im   = (const float*)d_in[7];
  const float* Dskip  = (const float*)d_in[8];
  const float* W_out  = (const float*)d_in[9];
  const float* b_out  = (const float*)d_in[10];
  const float* W_cls  = (const float*)d_in[11];
  const float* b_cls  = (const float*)d_in[12];
  float* out = (float*)d_out;

  // workspace carve-up (total ~800 KB)
  char* ws = (char*)d_ws;
  unsigned short* KtT = (unsigned short*)ws;                     // 128*384*2 = 98304
  float* Ksum     = (float*)(ws + 98304);                        // 384*128*4 = 196608
  float* node_sum = (float*)(ws + 98304 + 196608);               // 1024*128*4 = 524288

  s4_prep_kernel<<<(TPK * HH) / 256, 256, 0, stream>>>(
      log_dt, A_re, A_im, C_re, C_im, KtT, Ksum);
  s4_main_kernel<<<BNN, 256, 0, stream>>>(
      x, W_in, b_in, Dskip, W_out, b_out, KtT, Ksum, node_sum);
  s4_final_kernel<<<NB, 128, 0, stream>>>(node_sum, W_cls, b_cls, out);
}
